// TrainableMemory_1348619731446
// MI455X (gfx1250) — matrix operand order repaired and checked by reading, not compiled, not run on hardware
//
#include <hip/hip_runtime.h>
#include <hip/hip_bf16.h>
#include <math.h>

// Problem sizes (fixed by the reference).
#define B_Q   2048
#define D_DIM 512
#define M_K   65536
#define C_CLS 1000

#define MGROUPS 128                 // m-space split: 4096 tiles / 128 = 32 tiles per block
#define TILES_PER_BLOCK (M_K / 16 / MGROUPS)

typedef __attribute__((ext_vector_type(16))) _Float16 v16h;
typedef __attribute__((ext_vector_type(8)))  _Float16 v8h;
typedef __attribute__((ext_vector_type(8)))  float    v8f;

// Exact parameter type reported by clang for the async-LDS builtin:
typedef int v4i __attribute__((vector_size(4 * sizeof(int))));
#define AS1 __attribute__((address_space(1)))
#define AS3 __attribute__((address_space(3)))

// Order-preserving map float <-> uint32 (monotone).
__device__ __forceinline__ unsigned order_key(float f) {
    unsigned u = __float_as_uint(f);
    return (u & 0x80000000u) ? ~u : (u | 0x80000000u);
}
__device__ __forceinline__ float order_unkey(unsigned k) {
    return __uint_as_float((k & 0x80000000u) ? (k & 0x7FFFFFFFu) : ~k);
}

// --- gfx1250 async global->LDS staging helpers -----------------------------
#if defined(__has_builtin)
#  if __has_builtin(__builtin_amdgcn_global_load_async_to_lds_b128)
#    define HAVE_ASYNC_LDS 1
#  endif
#endif
#ifndef HAVE_ASYNC_LDS
#  define HAVE_ASYNC_LDS 0
#endif

// Stage one 16-row x 512-half key tile (16 KB, contiguous in k_f16) into LDS.
// 256 threads x 64 B each. Async path: 4x b128 async per thread (ASYNCcnt +=4).
__device__ __forceinline__ void stage_tile(const _Float16* __restrict__ kf,
                                           _Float16* dst, int m0, int tid) {
#if HAVE_ASYNC_LDS
    AS1 v4i* g = (AS1 v4i*)(kf + (size_t)m0 * D_DIM + tid * 32);
    AS3 v4i* l = (AS3 v4i*)(dst + tid * 32);
    __builtin_amdgcn_global_load_async_to_lds_b128(g,     l,     0, 0);
    __builtin_amdgcn_global_load_async_to_lds_b128(g + 1, l + 1, 0, 0);
    __builtin_amdgcn_global_load_async_to_lds_b128(g + 2, l + 2, 0, 0);
    __builtin_amdgcn_global_load_async_to_lds_b128(g + 3, l + 3, 0, 0);
#else
    const uint4* s = (const uint4*)(kf + (size_t)m0 * D_DIM);
    uint4* d = (uint4*)dst;
    #pragma unroll
    for (int i = 0; i < 4; ++i) d[tid + i * 256] = s[tid + i * 256];
#endif
}

__device__ __forceinline__ void wait_async_le4() {
#if HAVE_ASYNC_LDS
#  if __has_builtin(__builtin_amdgcn_s_wait_asynccnt)
    __builtin_amdgcn_s_wait_asynccnt(4);
#  else
    asm volatile("s_wait_asynccnt 0x4" ::: "memory");
#  endif
#endif
}
__device__ __forceinline__ void wait_async_le0() {
#if HAVE_ASYNC_LDS
#  if __has_builtin(__builtin_amdgcn_s_wait_asynccnt)
    __builtin_amdgcn_s_wait_asynccnt(0);
#  else
    asm volatile("s_wait_asynccnt 0x0" ::: "memory");
#  endif
#endif
}

// ---------------------------------------------------------------------------
// Kernel 1: L2-normalize rows (eps = 1e-12, matches F.normalize) -> f16.
// ---------------------------------------------------------------------------
__global__ void __launch_bounds__(256)
norm_rows_f16(const float* __restrict__ src, _Float16* __restrict__ dst) {
    const int row = blockIdx.x;
    const int tid = threadIdx.x;
    const float* s = src + (size_t)row * D_DIM;

    float ss = 0.f;
    #pragma unroll
    for (int i = 0; i < D_DIM / 256; ++i) {
        float v = s[tid + i * 256];
        ss += v * v;
    }
    #pragma unroll
    for (int m = 16; m >= 1; m >>= 1) ss += __shfl_xor(ss, m, 32);

    __shared__ float part[8];
    if ((tid & 31) == 0) part[tid >> 5] = ss;
    __syncthreads();
    float tot = 0.f;
    #pragma unroll
    for (int w = 0; w < 8; ++w) tot += part[w];

    const float scale = 1.0f / fmaxf(sqrtf(tot), 1e-12f);
    _Float16* d = dst + (size_t)row * D_DIM;
    #pragma unroll
    for (int i = 0; i < D_DIM / 256; ++i)
        d[tid + i * 256] = (_Float16)(s[tid + i * 256] * scale);
}

// ---------------------------------------------------------------------------
// Kernel 2: init packed (value,index) accumulators (0 < order_key(any float)).
// ---------------------------------------------------------------------------
__global__ void init_packed(unsigned long long* __restrict__ p) {
    int i = blockIdx.x * blockDim.x + threadIdx.x;
    if (i < B_Q) p[i] = 0ull;
}

// ---------------------------------------------------------------------------
// Kernel 3: fused sims = qn @ knT with running per-row (max, argmax).
// grid = (B_Q/128, MGROUPS), block = 256 (8 waves).
//   Each block owns 128 query rows; wave w owns rows b0 = bx*128 + w*16 and
//   keeps its 16 A-fragments (full D=512) in ~128 VGPRs.
//   All 8 waves sweep the SAME 32 m-tiles; the 16-row B tile (16 KB) is
//   double-buffered in LDS via async global->LDS copies and shared by all
//   waves -> 8x less L2 key traffic than per-wave streaming (1 GB total).
//
// A layout used (16x32): lane<16 holds K {d0+0..15} of row
// M=lane&15; lane>=16 holds K {d0+16..31} (matches B's contiguous order).
// B layout used (32x16): lane (g=lane>>4, n=lane&15) holds the 16
// contiguous halfs k[m0+n, d0+16g .. d0+16g+15]  (B = k^T, k row-major).
// ISA f32 C/D layout: elem r, lane l -> row r + 8*(l>>4), col l&15.
// ---------------------------------------------------------------------------
__global__ void __launch_bounds__(256)
sim_argmax(const _Float16* __restrict__ qf, const _Float16* __restrict__ kf,
           unsigned long long* __restrict__ packed) {
    __shared__ _Float16 kt[2][16 * D_DIM];       // 2 x 16 KB B-tile buffers

    const int tid  = threadIdx.x;
    const int lane = tid & 31;
    const int wave = tid >> 5;
    const int g    = lane >> 4;
    const int ln   = lane & 15;

    // ---- preload this wave's A fragments for all of D into registers ----
    const int b0 = blockIdx.x * 128 + wave * 16;
    const _Float16* arow = qf + (size_t)(b0 + ln) * D_DIM + g * 16;
    v16h afrag[16];
    #pragma unroll
    for (int s = 0; s < 16; ++s) {               // d0 = 32*s
        v8h alo = *(const v8h*)(arow + s * 32);
        v8h ahi = *(const v8h*)(arow + s * 32 + 8);
        afrag[s] = __builtin_shufflevector(alo, ahi,
                     0,1,2,3,4,5,6,7,8,9,10,11,12,13,14,15);
    }

    float    best[8];
    unsigned bcol[8];
    #pragma unroll
    for (int r = 0; r < 8; ++r) { best[r] = -INFINITY; bcol[r] = 0u; }

    const int tile0 = blockIdx.y * TILES_PER_BLOCK;

    // prime the pipeline
    stage_tile(kf, kt[0], tile0 * 16, tid);

    for (int t = 0; t < TILES_PER_BLOCK; ++t) {
        const int buf = t & 1;
        if (t + 1 < TILES_PER_BLOCK) {
            stage_tile(kf, kt[buf ^ 1], (tile0 + t + 1) * 16, tid);
            wait_async_le4();                    // older 4 asyncs (buf) done
        } else {
            wait_async_le0();
        }
        __syncthreads();                         // buf visible to all waves

        const int m0 = (tile0 + t) * 16;
        const _Float16* brow = kt[buf] + ln * D_DIM + g * 16;

        v8f c = {};
        #pragma unroll
        for (int s = 0; s < 16; ++s) {
            v16h bm = *(const v16h*)(brow + s * 32);   // 2x ds_load_b128
            c = __builtin_amdgcn_wmma_f32_16x16x32_f16(
                    false, afrag[s], false, bm, (short)0, c, false, false);
        }

        const unsigned col = (unsigned)(m0 + ln);
        #pragma unroll
        for (int r = 0; r < 8; ++r) {
            float v = c[r];
            if (v > best[r]) { best[r] = v; bcol[r] = col; } // strict >: first idx
        }
        __syncthreads();                         // done reading buf
    }

    // butterfly reduce within each 16-lane half
    #pragma unroll
    for (int r = 0; r < 8; ++r) {
        #pragma unroll
        for (int mm = 1; mm < 16; mm <<= 1) {
            float    ov = __shfl_xor(best[r], mm, 32);
            unsigned oc = (unsigned)__shfl_xor((int)bcol[r], mm, 32);
            if (ov > best[r] || (ov == best[r] && oc < bcol[r])) {
                best[r] = ov; bcol[r] = oc;
            }
        }
        if (ln == 0) {
            const int row = b0 + r + g * 8;
            // hi 32: value key; lo 32: ~col so ties prefer the smaller index.
            unsigned long long pk =
                ((unsigned long long)order_key(best[r]) << 32) |
                (unsigned long long)(~bcol[r]);
            atomicMax(packed + row, pk);
        }
    }
}

// ---------------------------------------------------------------------------
// Kernel 4: decode packed -> (confidence, m*), argmax over values[m*, :C).
// ---------------------------------------------------------------------------
__global__ void __launch_bounds__(256)
finalize(const unsigned long long* __restrict__ packed,
         const float* __restrict__ values,
         float* __restrict__ out_cls, float* __restrict__ out_conf) {
    const int b   = blockIdx.x;
    const int tid = threadIdx.x;

    const unsigned long long p = packed[b];
    const float    conf = order_unkey((unsigned)(p >> 32));
    const unsigned mi   = ~(unsigned)p;

    const float* row = values + (size_t)mi * C_CLS;
    float bv = -INFINITY; int bc = 0;
    for (int c = tid; c < C_CLS; c += 256) {
        float v = row[c];
        if (v > bv) { bv = v; bc = c; }
    }
    __shared__ float sv[256];
    __shared__ int   sc[256];
    sv[tid] = bv; sc[tid] = bc;
    __syncthreads();
    for (int s = 128; s > 0; s >>= 1) {
        if (tid < s) {
            if (sv[tid + s] > sv[tid] ||
                (sv[tid + s] == sv[tid] && sc[tid + s] < sc[tid])) {
                sv[tid] = sv[tid + s]; sc[tid] = sc[tid + s];
            }
        }
        __syncthreads();
    }
    if (tid == 0) {
        out_cls[b]  = (float)sc[0];
        out_conf[b] = conf;
    }
}

// ---------------------------------------------------------------------------
extern "C" void kernel_launch(void* const* d_in, const int* in_sizes, int n_in,
                              void* d_out, int out_size, void* d_ws, size_t ws_size,
                              hipStream_t stream) {
    const float* q = (const float*)d_in[0];   // [2048, 512]
    const float* k = (const float*)d_in[1];   // [65536, 512]
    const float* v = (const float*)d_in[2];   // [65536, 1000]

    float* out = (float*)d_out;               // [0,2048): classes  [2048,4096): conf

    char* ws = (char*)d_ws;
    _Float16* q_f16 = (_Float16*)ws;                                   //  2 MB
    _Float16* k_f16 = (_Float16*)(ws + (size_t)B_Q * D_DIM * 2);       // 64 MB
    unsigned long long* packed =
        (unsigned long long*)(ws + (size_t)B_Q * D_DIM * 2
                                 + (size_t)M_K * D_DIM * 2);           // 16 KB

    norm_rows_f16<<<B_Q, 256, 0, stream>>>(q, q_f16);
    norm_rows_f16<<<M_K, 256, 0, stream>>>(k, k_f16);
    init_packed<<<(B_Q + 255) / 256, 256, 0, stream>>>(packed);
    sim_argmax<<<dim3(B_Q / 128, MGROUPS), 256, 0, stream>>>(q_f16, k_f16, packed);
    finalize<<<B_Q, 256, 0, stream>>>(packed, v, out, out + B_Q);
}
